// MultiheadAttention_63419487093011
// MI455X (gfx1250) — matrix operand order repaired and checked by reading, not compiled, not run on hardware
//
#include <hip/hip_runtime.h>
#include <hip/hip_bf16.h>

// ---------------------------------------------------------------------------
// MultiheadAttention forward for MI455X (gfx1250, wave32, WMMA bf16).
// T=2048, B=2, E=1024, H=16, Dh=64.  All matmuls via v_wmma_f32_16x16x32_bf16.
// GEMMs double-buffer LDS with async global->LDS copies (ASYNCcnt path).
// ---------------------------------------------------------------------------

typedef __bf16 bf16_t;
typedef __bf16 v16bf __attribute__((ext_vector_type(16)));
typedef __bf16 v4bf  __attribute__((ext_vector_type(4)));
typedef float  v8f   __attribute__((ext_vector_type(8)));

union Frag { uint4 u[2]; v16bf v; };

#define WMMA_BF16(a, b, c) \
  __builtin_amdgcn_wmma_f32_16x16x32_bf16(false, (a), false, (b), (short)0, (c), false, false)

#if __has_builtin(__builtin_amdgcn_global_load_async_to_lds_b128)
#define ASYNC_COPY_OK 1
typedef int v4i __attribute__((ext_vector_type(4)));
typedef __attribute__((address_space(1))) v4i g_v4i;   // global int4
typedef __attribute__((address_space(3))) v4i l_v4i;   // LDS int4
#define ASYNC_LD128(gp, lp, off) \
  __builtin_amdgcn_global_load_async_to_lds_b128((g_v4i*)(gp), (l_v4i*)(lp), (off), 0)
#else
#define ASYNC_COPY_OK 0
#endif

#if __has_builtin(__builtin_amdgcn_s_wait_asynccnt)
#define WAIT_ASYNCCNT(N) __builtin_amdgcn_s_wait_asynccnt(N)
#else
#define WAIT_ASYNCCNT(N) asm volatile("s_wait_asynccnt " #N ::: "memory")
#endif

constexpr int EMBED = 1024;
constexpr int TSEQ  = 2048;
constexpr int BATCH = 2;
constexpr int HEADS = 16;
constexpr int HD    = 64;
constexpr int ROWS  = TSEQ * BATCH;   // 4096 flattened rows (r = t*B + b)
constexpr float NEG_MASK = -1e9f;

// ---------------------------------------------------------------------------
// fp32 -> bf16 elementwise convert (vectorized x4)
// ---------------------------------------------------------------------------
__global__ __launch_bounds__(256) void cvt_bf16_kernel(const float* __restrict__ in,
                                                       bf16_t* __restrict__ out) {
  size_t i = ((size_t)blockIdx.x * blockDim.x + threadIdx.x) * 4;
  float4 f = *(const float4*)(in + i);
  v4bf o;
  o.x = (bf16_t)f.x; o.y = (bf16_t)f.y; o.z = (bf16_t)f.z; o.w = (bf16_t)f.w;
  *(v4bf*)(out + i) = o;
}

// ---------------------------------------------------------------------------
// Weight transpose + convert: WT[n][k] = (bf16) W[k][n]   (1024x1024)
// ---------------------------------------------------------------------------
__global__ __launch_bounds__(256) void transpose_cvt_kernel(const float* __restrict__ W,
                                                            bf16_t* __restrict__ WT) {
  __shared__ float tile[32][33];
  int bx = blockIdx.x * 32;          // n-dim base
  int by = blockIdx.y * 32;          // k-dim base
  int tx = threadIdx.x, ty = threadIdx.y;   // block (32, 8)
#pragma unroll
  for (int i = 0; i < 32; i += 8)
    tile[ty + i][tx] = W[(size_t)(by + ty + i) * EMBED + (bx + tx)];
  __syncthreads();
#pragma unroll
  for (int i = 0; i < 32; i += 8)
    WT[(size_t)(bx + ty + i) * EMBED + (by + tx)] = (bf16_t)tile[tx][ty + i];
}

// ---------------------------------------------------------------------------
// Tiled WMMA GEMM:  C[4096 x 1024] = A[4096 x 1024] * WT^T  (+bias, epilogue)
//   Block tile 128x128, K-slice 64, double-buffered LDS filled by
//   GLOBAL_LOAD_ASYNC_TO_LDS_B128 (ASYNCcnt).  256 threads = 8 waves (2x4),
//   each wave computes 64x32 = 4x2 fragments, 16 WMMAs per K slice.
// MODE: 0 = Q  (bf16 out, (acc+bias)*scale)
//       1 = K  (bf16 out, acc+bias)
//       2 = V  (bf16 out scattered to Vt[bh][d][t])
//       3 = O  (fp32 out to d_out, acc+bias)
// ---------------------------------------------------------------------------
template <int MODE>
__global__ __launch_bounds__(256) void gemm_bf16_kernel(const bf16_t* __restrict__ A,
                                                        const bf16_t* __restrict__ WT,
                                                        const float* __restrict__ bias,
                                                        void* __restrict__ out,
                                                        float scale) {
  constexpr int BK  = 64;                     // K-slice width
  constexpr int LDP = 72;                     // padded LDS row stride (elements)
  __shared__ bf16_t lA[2][128 * LDP];
  __shared__ bf16_t lB[2][128 * LDP];

  const int blockN = blockIdx.x * 128;
  const int blockM = blockIdx.y * 128;
  const int tid    = threadIdx.x;
  const int lane   = tid & 31;
  const int waveId = tid >> 5;
  const int waveM  = waveId & 1;              // 2 wave-rows of 64
  const int waveN  = waveId >> 1;             // 4 wave-cols of 32
  const int ln     = lane & 15;
  const int lh     = lane >> 4;

  const int cpRow = tid >> 1;                 // 0..127
  const int cpSeg = tid & 1;                  // 32-element halves of 64-wide row

  const bf16_t* gA = A  + (size_t)(blockM + cpRow) * EMBED + cpSeg * 32;
  const bf16_t* gB = WT + (size_t)(blockN + cpRow) * EMBED + cpSeg * 32;
  const int sOff = cpRow * LDP + cpSeg * 32;  // LDS element offset of this thread's 64B

  // copy one 128xBK tile pair (A,B) for K offset k0 into LDS buffer `buf`
  auto issueTile = [&](int buf, int k0) {
    const bf16_t* ga = gA + k0;
    const bf16_t* gb = gB + k0;
    bf16_t* sa = &lA[buf][sOff];
    bf16_t* sb = &lB[buf][sOff];
#if ASYNC_COPY_OK
    ASYNC_LD128(ga,      sa,      0);
    ASYNC_LD128(ga + 8,  sa + 8,  0);
    ASYNC_LD128(ga + 16, sa + 16, 0);
    ASYNC_LD128(ga + 24, sa + 24, 0);
    ASYNC_LD128(gb,      sb,      0);
    ASYNC_LD128(gb + 8,  sb + 8,  0);
    ASYNC_LD128(gb + 16, sb + 16, 0);
    ASYNC_LD128(gb + 24, sb + 24, 0);
#else
#pragma unroll
    for (int q = 0; q < 4; ++q) {
      *(uint4*)(sa + q * 8) = *(const uint4*)(ga + q * 8);
      *(uint4*)(sb + q * 8) = *(const uint4*)(gb + q * 8);
    }
#endif
  };

  v8f acc[4][2] = {};

  issueTile(0, 0);
  constexpr int NITER = EMBED / BK;           // 16
  for (int it = 0; it < NITER; ++it) {
    const int buf = it & 1;
    if (it + 1 < NITER) {
      issueTile(buf ^ 1, (it + 1) * BK);      // prefetch next tile into other buffer
#if ASYNC_COPY_OK
      WAIT_ASYNCCNT(8);                       // current tile's 8 ops complete
#endif
    } else {
#if ASYNC_COPY_OK
      WAIT_ASYNCCNT(0);
#endif
    }
    __syncthreads();                          // all waves' copies visible

#pragma unroll
    for (int ks = 0; ks < 2; ++ks) {          // two 32-wide sub-slices
      Frag aF[4], bF[2];
#pragma unroll
      for (int mf = 0; mf < 4; ++mf) {
        // 16-bit A-fragment: half-wave K interleave {0-7,16-23}/{8-15,24-31}
        const bf16_t* r = &lA[buf][(waveM * 64 + mf * 16 + ln) * LDP + ks * 32 + lh * 8];
        aF[mf].u[0] = *(const uint4*)(r);
        aF[mf].u[1] = *(const uint4*)(r + 16);
      }
#pragma unroll
      for (int nf = 0; nf < 2; ++nf) {
        // B-fragment: lane = N column, half-wave K interleave {0-7,16-23}/{8-15,24-31}
        const bf16_t* r = &lB[buf][(waveN * 32 + nf * 16 + ln) * LDP + ks * 32 + lh * 8];
        bF[nf].u[0] = *(const uint4*)(r);
        bF[nf].u[1] = *(const uint4*)(r + 16);
      }
#pragma unroll
      for (int mf = 0; mf < 4; ++mf)
#pragma unroll
        for (int nf = 0; nf < 2; ++nf)
          acc[mf][nf] = WMMA_BF16(aF[mf].v, bF[nf].v, acc[mf][nf]);
    }
    __syncthreads();                          // reads done before buffer is refilled
  }

  // Epilogue. C layout: VGPR v -> M = v (+8 for upper half-wave), N = lane&15.
#pragma unroll
  for (int mf = 0; mf < 4; ++mf)
#pragma unroll
    for (int nf = 0; nf < 2; ++nf)
#pragma unroll
      for (int v = 0; v < 8; ++v) {
        int row = blockM + waveM * 64 + mf * 16 + v + lh * 8;
        int col = blockN + waveN * 32 + nf * 16 + ln;
        float val = acc[mf][nf][v] + bias[col];
        if (MODE == 0) {
          ((bf16_t*)out)[(size_t)row * EMBED + col] = (bf16_t)(val * scale);
        } else if (MODE == 1) {
          ((bf16_t*)out)[(size_t)row * EMBED + col] = (bf16_t)val;
        } else if (MODE == 2) {
          int t = row >> 1, bb = row & 1;          // row = t*BATCH + b
          int hh = col >> 6, dd = col & 63;        // col = h*64 + d
          ((bf16_t*)out)[(size_t)((bb * HEADS + hh) * HD + dd) * TSEQ + t] = (bf16_t)val;
        } else {
          ((float*)out)[(size_t)row * EMBED + col] = val;
        }
      }
}

// ---------------------------------------------------------------------------
// Flash attention: grid = (T/64 row-blocks, B*H).  128 threads = 4 waves,
// each wave owns 16 query rows.  Online softmax in fp32, QK^T and PV via WMMA.
// ---------------------------------------------------------------------------
__global__ __launch_bounds__(128) void attn_kernel(const bf16_t* __restrict__ Qp,
                                                   const bf16_t* __restrict__ Kp,
                                                   const bf16_t* __restrict__ Vt,
                                                   const int* __restrict__ kpm,
                                                   bf16_t* __restrict__ Aout) {
  const int bh   = blockIdx.y;
  const int b    = bh >> 4;
  const int h    = bh & 15;
  const int wave = threadIdx.x >> 5;
  const int lane = threadIdx.x & 31;
  const int ln   = lane & 15;
  const int lh   = lane >> 4;
  const int tBase = blockIdx.x * 64 + wave * 16;

  __shared__ bf16_t Plds[4][16 * 64];   // per-wave P tile, row-major 16x64

  // Q A-fragments over d (K=64 -> two 32-slices), interleaved A layout
  Frag aQ[2];
#pragma unroll
  for (int dsl = 0; dsl < 2; ++dsl) {
    const bf16_t* qr = Qp + ((size_t)(tBase + ln) * BATCH + b) * EMBED + h * HD + dsl * 32;
    aQ[dsl].u[0] = *(const uint4*)(qr + lh * 8);
    aQ[dsl].u[1] = *(const uint4*)(qr + lh * 8 + 16);
  }

  v8f o[4] = {};
  float m[8], lsum[8];
#pragma unroll
  for (int v = 0; v < 8; ++v) { m[v] = -__builtin_inff(); lsum[v] = 0.f; }

  const int nTiles = blockIdx.x + 1;   // causal: only sBase <= tMax of block
  for (int js = 0; js < nTiles; ++js) {
    const int sBase = js * 64;

    // prefetch next S tile's K rows and V^T rows while we crunch this one
    if (js + 1 < nTiles) {
      const int sn = (js + 1) * 64;
      __builtin_prefetch(Kp + ((size_t)(sn + lane) * BATCH + b) * EMBED + h * HD, 0, 1);
      __builtin_prefetch(Kp + ((size_t)(sn + 32 + lane) * BATCH + b) * EMBED + h * HD, 0, 1);
      __builtin_prefetch(Vt + ((size_t)bh * HD + lane) * TSEQ + sn, 0, 1);
      __builtin_prefetch(Vt + ((size_t)bh * HD + 32 + lane) * TSEQ + sn, 0, 1);
    }

    // ---- S = Q K^T  (16 x 64 scores in 4 fp32 fragments) ----
    v8f s[4] = {};
#pragma unroll
    for (int dsl = 0; dsl < 2; ++dsl) {
#pragma unroll
      for (int nt = 0; nt < 4; ++nt) {
        Frag bK;
        const int scol = sBase + nt * 16 + ln;
        const bf16_t* kr = Kp + ((size_t)scol * BATCH + b) * EMBED + h * HD + dsl * 32;
        bK.u[0] = *(const uint4*)(kr + lh * 8);
        bK.u[1] = *(const uint4*)(kr + lh * 8 + 16);
        s[nt] = WMMA_BF16(aQ[dsl].v, bK.v, s[nt]);
      }
    }

    // ---- masks + online softmax (C layout: row = v + 8*lh, col = nt*16+ln) ----
    int pad[4];
#pragma unroll
    for (int nt = 0; nt < 4; ++nt) pad[nt] = kpm[b * TSEQ + sBase + nt * 16 + ln];

    float mnew[8];
#pragma unroll
    for (int v = 0; v < 8; ++v) {
      const int t = tBase + v + lh * 8;
      float mx = m[v];
#pragma unroll
      for (int nt = 0; nt < 4; ++nt) {
        const int scol = sBase + nt * 16 + ln;
        float e = s[nt][v];
        e += (scol <= t) ? 0.f : NEG_MASK;         // additive causal mask
        if (pad[nt]) e = -__builtin_inff();        // key padding mask
        s[nt][v] = e;
        mx = fmaxf(mx, e);
      }
#pragma unroll
      for (int off = 1; off < 16; off <<= 1)       // row max across 16-lane half
        mx = fmaxf(mx, __shfl_xor(mx, off, 32));
      mnew[v] = mx;
    }

#pragma unroll
    for (int v = 0; v < 8; ++v) {
      const float sc = __expf(m[v] - mnew[v]);     // first tile: m=-inf, mnew finite
      m[v] = mnew[v];
      lsum[v] *= sc;
#pragma unroll
      for (int nt = 0; nt < 4; ++nt) o[nt][v] *= sc;
    }

    // P = exp(s - m); accumulate row sums; spill P (bf16) to LDS for re-layout
#pragma unroll
    for (int v = 0; v < 8; ++v) {
      float sum = 0.f;
#pragma unroll
      for (int nt = 0; nt < 4; ++nt) {
        const float pv = __expf(s[nt][v] - m[v]);
        Plds[wave][(v + lh * 8) * 64 + nt * 16 + ln] = (bf16_t)pv;
        sum += pv;
      }
#pragma unroll
      for (int off = 1; off < 16; off <<= 1)
        sum += __shfl_xor(sum, off, 32);
      lsum[v] += sum;
    }

    // ---- O += P V  (P re-read as A-fragments; V^T rows are contiguous) ----
    Frag aP[2];
#pragma unroll
    for (int kf = 0; kf < 2; ++kf) {
      const bf16_t* pr = &Plds[wave][ln * 64 + kf * 32];
      aP[kf].u[0] = *(const uint4*)(pr + lh * 8);
      aP[kf].u[1] = *(const uint4*)(pr + lh * 8 + 16);
    }
#pragma unroll
    for (int kf = 0; kf < 2; ++kf) {
#pragma unroll
      for (int nt = 0; nt < 4; ++nt) {
        Frag bV;
        const int d = nt * 16 + ln;
        const bf16_t* vr = Vt + ((size_t)bh * HD + d) * TSEQ + sBase + kf * 32;
        bV.u[0] = *(const uint4*)(vr + lh * 8);
        bV.u[1] = *(const uint4*)(vr + lh * 8 + 16);
        o[nt] = WMMA_BF16(aP[kf].v, bV.v, o[nt]);
      }
    }
  }

  // normalize and write attention output (bf16, [row=t*B+b][h*64+d])
#pragma unroll
  for (int v = 0; v < 8; ++v) {
    const float inv = 1.f / lsum[v];
    const int t = tBase + v + lh * 8;
#pragma unroll
    for (int nt = 0; nt < 4; ++nt)
      Aout[((size_t)t * BATCH + b) * EMBED + h * HD + nt * 16 + ln] = (bf16_t)(o[nt][v] * inv);
  }
}

// ---------------------------------------------------------------------------
// Launcher
// ---------------------------------------------------------------------------
extern "C" void kernel_launch(void* const* d_in, const int* in_sizes, int n_in,
                              void* d_out, int out_size, void* d_ws, size_t ws_size,
                              hipStream_t stream) {
  const float* query = (const float*)d_in[0];
  const float* key_t = (const float*)d_in[1];
  const float* value = (const float*)d_in[2];
  // d_in[3] = attn_mask (causal, computed analytically in-kernel)
  const int*   kpm   = (const int*)d_in[4];
  const float* Wq = (const float*)d_in[5];  const float* bq = (const float*)d_in[6];
  const float* Wk = (const float*)d_in[7];  const float* bk = (const float*)d_in[8];
  const float* Wv = (const float*)d_in[9];  const float* bv = (const float*)d_in[10];
  const float* Wo = (const float*)d_in[11]; const float* bo = (const float*)d_in[12];

  const size_t X_BYTES = (size_t)ROWS * EMBED * sizeof(bf16_t);   // 8 MB
  const size_t W_BYTES = (size_t)EMBED * EMBED * sizeof(bf16_t);  // 2 MB

  char* p = (char*)d_ws;
  auto take = [&](size_t bytes) { void* r = (void*)p; p += (bytes + 255) & ~(size_t)255; return r; };
  bf16_t* Xq  = (bf16_t*)take(X_BYTES);
  bf16_t* Xk  = (bf16_t*)take(X_BYTES);
  bf16_t* Xv  = (bf16_t*)take(X_BYTES);
  bf16_t* WqT = (bf16_t*)take(W_BYTES);
  bf16_t* WkT = (bf16_t*)take(W_BYTES);
  bf16_t* WvT = (bf16_t*)take(W_BYTES);
  bf16_t* WoT = (bf16_t*)take(W_BYTES);
  bf16_t* Qp  = (bf16_t*)take(X_BYTES);
  bf16_t* Kp  = (bf16_t*)take(X_BYTES);
  bf16_t* Vt  = (bf16_t*)take(X_BYTES);   // [B*H][64][2048] d-major
  bf16_t* Ao  = (bf16_t*)take(X_BYTES);

  const float scaling = 0.125f;           // 64^-0.5

  // 1) bf16 conversions
  const int cvtBlocks = ROWS * EMBED / 4 / 256;   // 4096
  cvt_bf16_kernel<<<cvtBlocks, 256, 0, stream>>>(query, Xq);
  cvt_bf16_kernel<<<cvtBlocks, 256, 0, stream>>>(key_t, Xk);
  cvt_bf16_kernel<<<cvtBlocks, 256, 0, stream>>>(value, Xv);

  dim3 tb(32, 8), tg(EMBED / 32, EMBED / 32);
  transpose_cvt_kernel<<<tg, tb, 0, stream>>>(Wq, WqT);
  transpose_cvt_kernel<<<tg, tb, 0, stream>>>(Wk, WkT);
  transpose_cvt_kernel<<<tg, tb, 0, stream>>>(Wv, WvT);
  transpose_cvt_kernel<<<tg, tb, 0, stream>>>(Wo, WoT);

  // 2) projection GEMMs (WMMA bf16, fp32 accumulate, async-LDS double buffer)
  dim3 gg(EMBED / 128, ROWS / 128);       // (8, 32)
  gemm_bf16_kernel<0><<<gg, 256, 0, stream>>>(Xq, WqT, bq, (void*)Qp, scaling);
  gemm_bf16_kernel<1><<<gg, 256, 0, stream>>>(Xk, WkT, bk, (void*)Kp, 1.f);
  gemm_bf16_kernel<2><<<gg, 256, 0, stream>>>(Xv, WvT, bv, (void*)Vt, 1.f);

  // 3) flash attention
  dim3 ag(TSEQ / 64, BATCH * HEADS);      // (32, 32)
  attn_kernel<<<ag, 128, 0, stream>>>(Qp, Kp, Vt, kpm, Ao);

  // 4) output projection -> fp32 d_out
  gemm_bf16_kernel<3><<<gg, 256, 0, stream>>>(Ao, WoT, bo, d_out, 1.f);
}
